// LFQQuantizer_44306882625715
// MI455X (gfx1250) — matrix operand order repaired and checked by reading, not compiled, not run on hardware
//
#include <hip/hip_runtime.h>
#include <hip/hip_bf16.h>

typedef __attribute__((ext_vector_type(16))) _Float16      v16h;
typedef __attribute__((ext_vector_type(8)))  float         v8f;
typedef __attribute__((ext_vector_type(8)))  unsigned int  v8u;

#define D_DIM        14
#define K_CODES      16384
#define N_SAMPLES    8192          // 8*1024
#define COL_TILES    1024          // 16384/16
#define ROW_TILES    512           // 8192/16
#define WAVES        8
#define THREADS      256
#define SC2          288.539008f   // (2/TEMPERATURE) * log2(e) = 200/ln2
#define LN2_F        0.69314718056f
#define EPS_F        1e-5f
#define Q_ELEMS      114688        // 8*1024*14

// workspace (floats): [0..16383] avg_probs accum, [16384] sample-entropy accum (nats),
// [16385] commit-loss accum
#define WS_AVG        0
#define WS_SENT       16384
#define WS_COMMIT     16385
#define WS_FLOATS     16386

// raw hardware transcendentals: v_exp_f32 is exp2, v_log_f32 is log2
#define EXP2F(x) __builtin_amdgcn_exp2f(x)
#define LOG2F(x) __builtin_amdgcn_logf(x)

__device__ __forceinline__ float hsum16(float v) {
#pragma unroll
    for (int off = 1; off < 16; off <<= 1)
        v += __shfl_xor(v, off, 32);
    return v;
}

// pack two codebook bits (k, k+1) of v into an f16 pair (+1.0h / -1.0h)
__device__ __forceinline__ unsigned pk16(int v, int k) {
    unsigned lo = ((v >> k) & 1)       ? 0x3C00u : 0xBC00u;
    unsigned hi = ((v >> (k + 1)) & 1) ? 0x3C00u : 0xBC00u;
    return lo | (hi << 16);
}

// B fragment for column tile t (scalar) given the per-lane constant pieces
__device__ __forceinline__ v16h make_b(int t, unsigned w0, unsigned w1, unsigned lmask) {
    v8u bw;
    bw[0] = w0;                 bw[1] = w1;
    bw[2] = pk16(t, 0) & lmask; bw[3] = pk16(t, 2) & lmask;
    bw[4] = pk16(t, 4) & lmask; bw[5] = pk16(t, 6) & lmask;
    bw[6] = pk16(t, 8) & lmask; bw[7] = 0u;
    return __builtin_bit_cast(v16h, bw);
}

__global__ void lfq_init_ws(float* ws) {
    int i = blockIdx.x * blockDim.x + threadIdx.x;
    if (i < WS_FLOATS) ws[i] = 0.0f;
}

// Fused logits (WMMA) + factored softmax statistics + entropies.
__global__ void __launch_bounds__(THREADS)
lfq_entropy_kernel(const float* __restrict__ x, float* __restrict__ ws) {
    __shared__ float xs[16 * 16];     // 16 samples x 16 dims (padded)
    __shared__ float sl[WAVES][16];   // per-wave per-row partial sums

    const int tid  = threadIdx.x;
    // wave id is uniform within the wave: pin it to an SGPR so the tile loop
    // counter and the tile-dependent codebook bits stay scalar (SALU + s_cbranch)
    const int wave = __builtin_amdgcn_readfirstlane(tid >> 5);
    const int lane = tid & 31;
    const int half = (lane < 16) ? 0 : 8;

    // stage x tile into LDS (zero-pad dims 14,15)
    for (int i = tid; i < 256; i += THREADS) {
        int r = i >> 4, c = i & 15;
        xs[i] = (c < D_DIM) ? x[(blockIdx.x * 16 + r) * D_DIM + c] : 0.0f;
    }
    __syncthreads();

    // A fragment (16x32 f16): lane m = lane&15 is the row.
    // elements 0-15 -> K 0-15 (lanes<16) / 16-31 (lanes>=16).
    v16h a = {};
    {
        const int m = lane & 15;
#pragma unroll
        for (int e = 0; e < 16; ++e) {
            int k = (lane < 16) ? e : e + 16;
            a[e] = (_Float16)((k < D_DIM) ? xs[m * 16 + k] : 0.0f);
        }
    }

    // Exact per-row softmax max: max_j (x . c_j) == ||x_row||_1, in log2 domain.
    float M2[8];
#pragma unroll
    for (int r = 0; r < 8; ++r) {
        const int row = r + half;
        float s1 = 0.0f;
#pragma unroll
        for (int k = 0; k < D_DIM; ++k) s1 += fabsf(xs[row * 16 + k]);
        M2[r] = SC2 * s1;
    }

    // Lane-invariant pieces of the B fragment (code j = 16*t + lane):
    // elements 0-3 come from lane bits; lanes >=16 hold K=16..31 which are all zero.
    const unsigned lmask = (lane < 16) ? 0xFFFFFFFFu : 0u;
    const unsigned w0 = pk16(lane, 0) & lmask;
    const unsigned w1 = pk16(lane, 2) & lmask;

    // ---------------- pass 1: denominator sums (max is analytic) ----------------
    float l[8];
#pragma unroll
    for (int r = 0; r < 8; ++r) l[r] = 0.0f;

#pragma unroll 2
    for (int t = wave; t < COL_TILES; t += WAVES) {
        v16h b = make_b(t, w0, w1, lmask);
        v8f c = {};
        c = __builtin_amdgcn_wmma_f32_16x16x32_f16(false, a, false, b,
                                                   (short)0, c, false, false);
#pragma unroll
        for (int r = 0; r < 8; ++r)
            l[r] += EXP2F(__fmaf_rn(SC2, c[r], -M2[r]));
    }

    // single end-of-pass reduction: 16 lanes, then 8 waves via LDS
#pragma unroll
    for (int r = 0; r < 8; ++r) l[r] = hsum16(l[r]);
    if (lane == 0) {
#pragma unroll
        for (int r = 0; r < 8; ++r) sl[wave][r] = l[r];
    }
    if (lane == 16) {
#pragma unroll
        for (int r = 0; r < 8; ++r) sl[wave][r + 8] = l[r];
    }
    __syncthreads();

    float invL[8], lg2L[8];
#pragma unroll
    for (int r = 0; r < 8; ++r) {
        const int row = r + half;
        float Ls = 0.0f;
#pragma unroll
        for (int w = 0; w < WAVES; ++w) Ls += sl[w][row];
        invL[r] = 1.0f / Ls;
        lg2L[r] = LOG2F(Ls);
    }

    // ---------------- pass 2: probs, entropies, avg_probs ----------------
    float ent[8];
#pragma unroll
    for (int r = 0; r < 8; ++r) ent[r] = 0.0f;

    float* const avg_slot = ws + WS_AVG + (lane & 15);

#pragma unroll 2
    for (int t = wave; t < COL_TILES; t += WAVES) {
        v16h b = make_b(t, w0, w1, lmask);
        v8f c = {};
        c = __builtin_amdgcn_wmma_f32_16x16x32_f16(false, a, false, b,
                                                   (short)0, c, false, false);
        float pcol = 0.0f;          // sum of p over this lane's 8 rows (one column)
#pragma unroll
        for (int r = 0; r < 8; ++r) {
            float u = __fmaf_rn(SC2, c[r], -M2[r]);      // s2 - max2
            float p = EXP2F(u) * invL[r];
            ent[r]  = __fmaf_rn(p, lg2L[r] - u, ent[r]); // -p*log2(p)
            pcol   += p;
        }
        // both half-waves hit the same 16 slots; non-returning f32 atomics, no ds fold
        atomicAdd(avg_slot + t * 16, pcol);
    }

    // wave-level sample-entropy partial (bits -> nats), one reduction at the end
    float etot = 0.0f;
#pragma unroll
    for (int r = 0; r < 8; ++r) etot += hsum16(ent[r]);
    if ((lane & 15) == 0)
        atomicAdd(&ws[WS_SENT], etot * LN2_F);
}

__global__ void lfq_sign_commit(const float* __restrict__ x,
                                float* __restrict__ out,
                                float* __restrict__ ws, int n) {
    int i = blockIdx.x * blockDim.x + threadIdx.x;
    float acc = 0.0f;
    if (i < n) {
        float v = x[i];
        float q = (v > 0.0f) ? 1.0f : -1.0f;
        out[i]  = q;
        float d = v - q;
        acc = d * d;
    }
#pragma unroll
    for (int off = 1; off < 32; off <<= 1)
        acc += __shfl_xor(acc, off, 32);
    if ((threadIdx.x & 31) == 0)
        atomicAdd(&ws[WS_COMMIT], acc);
}

__global__ void lfq_finalize(const float* __restrict__ ws, float* __restrict__ out) {
    __shared__ float red[THREADS];
    const int tid = threadIdx.x;
    float acc = 0.0f;
    for (int j = tid; j < K_CODES; j += THREADS) {
        float p = ws[WS_AVG + j] * (1.0f / (float)N_SAMPLES);
        acc += -p * __logf(p + EPS_F);
    }
    red[tid] = acc;
    __syncthreads();
    for (int s = THREADS / 2; s > 0; s >>= 1) {
        if (tid < s) red[tid] += red[tid + s];
        __syncthreads();
    }
    if (tid == 0) {
        float avg_entropy    = red[0];
        float sample_entropy = ws[WS_SENT] * (1.0f / (float)N_SAMPLES);
        float commit_loss    = ws[WS_COMMIT] * (1.0f / (float)Q_ELEMS);
        out[Q_ELEMS + 0] = sample_entropy - avg_entropy;  // entropy_aux_loss
        out[Q_ELEMS + 1] = sample_entropy;
        out[Q_ELEMS + 2] = avg_entropy;
        out[Q_ELEMS + 3] = commit_loss;
    }
}

extern "C" void kernel_launch(void* const* d_in, const int* in_sizes, int n_in,
                              void* d_out, int out_size, void* d_ws, size_t ws_size,
                              hipStream_t stream) {
    (void)in_sizes; (void)n_in; (void)out_size; (void)ws_size;
    const float* x  = (const float*)d_in[0];
    float* out      = (float*)d_out;
    float* ws       = (float*)d_ws;

    lfq_init_ws<<<(WS_FLOATS + 255) / 256, 256, 0, stream>>>(ws);
    lfq_entropy_kernel<<<ROW_TILES, THREADS, 0, stream>>>(x, ws);
    lfq_sign_commit<<<(Q_ELEMS + 255) / 256, 256, 0, stream>>>(x, out, ws, Q_ELEMS);
    lfq_finalize<<<1, THREADS, 0, stream>>>(ws, out);
}
